// CustomBlock_6511170421768
// MI455X (gfx1250) — matrix operand order repaired and checked by reading, not compiled, not run on hardware
//
#include <hip/hip_runtime.h>

typedef unsigned short u16;
typedef unsigned int   u32;

typedef __attribute__((ext_vector_type(16))) __bf16 v16bf;
typedef __attribute__((ext_vector_type(8)))  float  v8f;
typedef __attribute__((ext_vector_type(4)))  int    v4i;

union FragBF { uint4 q[2]; v16bf v; };

__device__ __forceinline__ u16 f2bf(float f) {
  u32 u = __float_as_uint(f);
  u += 0x7FFFu + ((u >> 16) & 1u);   // round-to-nearest-even
  return (u16)(u >> 16);
}

// ---- async global->LDS copy (CDNA5), with synchronous fallback ----
#if defined(__has_builtin)
#if __has_builtin(__builtin_amdgcn_global_load_async_to_lds_b128)
#define HAVE_ASYNC_LDS 1
#endif
#endif

#ifdef HAVE_ASYNC_LDS
typedef __attribute__((address_space(1))) v4i gas_v4i;   // global int4
typedef __attribute__((address_space(3))) v4i las_v4i;   // LDS int4
#define ASYNC_B128(g, l)                                                       \
  __builtin_amdgcn_global_load_async_to_lds_b128(                              \
      (gas_v4i*)(g), (las_v4i*)(l), 0, 0)
__device__ __forceinline__ void await_async() {
#if __has_builtin(__builtin_amdgcn_s_wait_asynccnt)
  __builtin_amdgcn_s_wait_asynccnt(0);
#else
  asm volatile("s_wait_asynccnt 0" ::: "memory");
#endif
}
#else
#define ASYNC_B128(g, l) (*(uint4*)(l) = *(const uint4*)(g))
__device__ __forceinline__ void await_async() {}
#endif

// ---------------- weight pack: [Cout][Cin][3][3] f32 -> [Cout][(dh*3+dw)*CIN+cin] bf16
template<int CIN>
__global__ __launch_bounds__(256) void pack_w_kernel(const float* __restrict__ w,
                                                     u16* __restrict__ out) {
  constexpr int K = CIN * 9;
  int idx = blockIdx.x * 256 + threadIdx.x;
  if (idx >= 128 * K) return;
  int co = idx / K;
  int k  = idx - co * K;
  int g  = k / CIN;          // dh*3+dw
  int ci = k - g * CIN;
  out[idx] = f2bf(w[(size_t)(co * CIN + ci) * 9 + g]);
}

// ---------------- x: NCHW f32 -> NHWC bf16 (Cin=64, H=W=112), one (b,h) row per block
__global__ __launch_bounds__(256) void to_nhwc_kernel(const float* __restrict__ x,
                                                      u16* __restrict__ out) {
  __shared__ u16 lds[112 * 64];
  int b = blockIdx.x / 112;
  int h = blockIdx.x - b * 112;
  const float* src = x + (size_t)b * 64 * 12544 + (size_t)h * 112;
  for (int i = threadIdx.x; i < 7168; i += 256) {
    int c = i / 112;
    int w = i - c * 112;
    lds[w * 64 + c] = f2bf(src[(size_t)c * 12544 + w]);
  }
  __syncthreads();
  u32* dst = (u32*)(out + (size_t)blockIdx.x * 7168);
  const u32* s = (const u32*)lds;
  for (int i = threadIdx.x; i < 3584; i += 256) dst[i] = s[i];
}

// ---------------- implicit-GEMM 3x3 conv, bf16 WMMA, optional fused 2x2 maxpool
// Block tile: (4 rows x 16 cols) spatial x 128 out-channels; 8 waves, each 16Mx64N.
// LDS double-buffered in K=64 stages (2 WMMA k-steps per stage), filled with
// async global->LDS copies (ASYNCcnt), so no VGPR round-trip for tile staging.
template<int CIN, bool POOL>
__global__ __launch_bounds__(256) void conv3x3_wmma_kernel(
    const u16* __restrict__ X,     // bf16 NHWC [B][112][112][CIN]
    const u16* __restrict__ Wp,    // bf16 [128][CIN*9]
    const float* __restrict__ bias,
    u16* __restrict__ Y,           // bf16 NHWC [B][112][112][128]   (!POOL)
    float* __restrict__ P)         // f32  NCHW [B][128][56][56]     (POOL)
{
  constexpr int K    = CIN * 9;
  constexpr int ST   = K / 64;     // K stages (64-wide; tap-aligned since 64 | CIN)
  constexpr int ASTR = 72;         // halves per A row (64 + 8 pad)
  constexpr int BSTR = 72;         // halves per B row

  __shared__ __align__(16) unsigned char smem[55296];
  u16*   Alds = (u16*)smem;             // [2][64][72]  = 18432 B
  u16*   Blds = (u16*)(smem + 18432);   // [2][128][72] = 36864 B
  float* Olds = (float*)smem;           // [64][144]    = 36864 B (reused after K loop)

  const int b  = blockIdx.z;
  const int h0 = blockIdx.y * 4;
  const int w0 = blockIdx.x * 16;

  const int tid  = threadIdx.x;
  const int lane = tid & 31;
  const int wave = tid >> 5;
  const int wm   = wave & 3;     // 16-row M sub-tile (spatial row in tile)
  const int wn   = wave >> 2;    // 64-channel N half
  const int col  = lane & 15;
  const int kh   = lane >> 4;

  // cooperative-load coordinates
  const int am  = tid >> 2;      // A row 0..63
  const int aq  = tid & 3;       // 16-half quarter of the 64-half row
  const int alh = am >> 4;
  const int alw = am & 15;
  const int bch = tid >> 1;      // B row (out channel) 0..127
  const int bq  = tid & 1;       // 32-half half of the 64-half row

  v8f acc[4];
#pragma unroll
  for (int jt = 0; jt < 4; ++jt)
#pragma unroll
    for (int r = 0; r < 8; ++r) acc[jt][r] = 0.0f;

  float bv[4];
#pragma unroll
  for (int jt = 0; jt < 4; ++jt) bv[jt] = bias[wn * 64 + jt * 16 + col];

  auto stageLoad = [&](int st, int buf) {
    int k0  = st * 64;
    int g   = k0 / CIN;          // tap index dh*3+dw (64-slice never crosses a tap)
    int ci0 = k0 & (CIN - 1);
    int dh  = g / 3;
    int dw  = g - dh * 3;
    int hh  = h0 + alh + dh - 1;
    int ww  = w0 + alw + dw - 1;
    u16* al = &Alds[buf * 64 * ASTR + am * ASTR + aq * 16];
    if ((unsigned)hh < 112u && (unsigned)ww < 112u) {
      const u16* gp = X + (((size_t)b * 112 + hh) * 112 + ww) * CIN + ci0 + aq * 16;
      ASYNC_B128(gp, al);
      ASYNC_B128(gp + 8, al + 8);
    } else {
      uint4 z = make_uint4(0u, 0u, 0u, 0u);
      *(uint4*)al = z;
      *(uint4*)(al + 8) = z;
    }
    const u16* wp = Wp + (size_t)bch * K + k0 + bq * 32;
    u16* bl = &Blds[buf * 128 * BSTR + bch * BSTR + bq * 32];
    ASYNC_B128(wp,      bl);
    ASYNC_B128(wp + 8,  bl + 8);
    ASYNC_B128(wp + 16, bl + 16);
    ASYNC_B128(wp + 24, bl + 24);
  };

  stageLoad(0, 0);
  await_async();
  __syncthreads();

  for (int st = 0; st < ST; ++st) {
    const int cur = st & 1;
    const int nxt = cur ^ 1;
    if (st + 1 < ST) stageLoad(st + 1, nxt);   // async into other buffer

    const u16* abase = &Alds[cur * 64 * ASTR + (wm * 16 + col) * ASTR];
#pragma unroll
    for (int s = 0; s < 2; ++s) {
      // A fragment: 16-bit 16x32 layout (lane<16: K{0..7,16..23}; lane>=16: K{8..15,24..31})
      FragBF fa;
      fa.q[0] = *(const uint4*)(abase + s * 32 + kh * 8);
      fa.q[1] = *(const uint4*)(abase + s * 32 + 16 + kh * 8);
#pragma unroll
      for (int jt = 0; jt < 4; ++jt) {
        // B fragment: lane holds column N=col, K{8*kh..8*kh+7, 16+8*kh..16+8*kh+7} (N-major tile)
        FragBF fb;
        const u16* bbase =
            &Blds[cur * 128 * BSTR + (wn * 64 + jt * 16 + col) * BSTR + s * 32 + kh * 8];
        fb.q[0] = *(const uint4*)bbase;
        fb.q[1] = *(const uint4*)(bbase + 16);
        acc[jt] = __builtin_amdgcn_wmma_f32_16x16x32_bf16(
            false, fa.v, false, fb.v, (short)0, acc[jt], false, false);
      }
    }

    await_async();      // next-stage tiles resident in LDS
    __syncthreads();
  }

  // epilogue: bias + ReLU into LDS (stride 144 avoids bank conflicts)
#pragma unroll
  for (int jt = 0; jt < 4; ++jt) {
    int ch = wn * 64 + jt * 16 + col;
#pragma unroll
    for (int r = 0; r < 8; ++r) {
      float v = acc[jt][r] + bv[jt];
      Olds[(wm * 16 + kh * 8 + r) * 144 + ch] = fmaxf(v, 0.0f);
    }
  }
  __syncthreads();

  if (!POOL) {
    // coalesced NHWC bf16 store: each thread packs 32 channels of one pixel
    const int m   = tid >> 2;
    const int seg = tid & 3;
    const int lh  = m >> 4, lw = m & 15;
    u16* dst = Y + (((size_t)b * 112 + (h0 + lh)) * 112 + (w0 + lw)) * 128 + seg * 32;
    const float* src = &Olds[m * 144 + seg * 32];
#pragma unroll
    for (int q = 0; q < 4; ++q) {
      uint4 pk;
      u16* pp = (u16*)&pk;
#pragma unroll
      for (int e = 0; e < 8; ++e) pp[e] = f2bf(src[q * 8 + e]);
      *(uint4*)(dst + q * 8) = pk;
    }
  } else {
    // fused 2x2 maxpool -> f32 NCHW output
#pragma unroll
    for (int i = 0; i < 8; ++i) {
      int idx = i * 256 + tid;     // 2048 pooled outputs
      int ch  = idx >> 4;
      int rem = idx & 15;
      int phl = rem >> 3;
      int pwl = rem & 7;
      int r0  = (phl * 2) * 16 + pwl * 2;
      int r1  = r0 + 16;
      float v = fmaxf(fmaxf(Olds[r0 * 144 + ch], Olds[(r0 + 1) * 144 + ch]),
                      fmaxf(Olds[r1 * 144 + ch], Olds[(r1 + 1) * 144 + ch]));
      int ph = (h0 >> 1) + phl;
      int pw = (w0 >> 1) + pwl;
      P[(((size_t)b * 128 + ch) * 56 + ph) * 56 + pw] = v;
    }
  }
}

extern "C" void kernel_launch(void* const* d_in, const int* in_sizes, int n_in,
                              void* d_out, int out_size, void* d_ws, size_t ws_size,
                              hipStream_t stream) {
  const float* x  = (const float*)d_in[0];
  const float* w1 = (const float*)d_in[1];
  const float* b1 = (const float*)d_in[2];
  const float* w2 = (const float*)d_in[3];
  const float* b2 = (const float*)d_in[4];

  char* ws = (char*)d_ws;
  u16* W1p = (u16*)ws;                                  // 147456 B
  u16* W2p = (u16*)(ws + 147456);                       // 294912 B
  u16* Xh  = (u16*)(ws + 147456 + 294912);              // 51380224 B  (x as NHWC bf16)
  u16* Y1  = (u16*)(ws + 147456 + 294912 + 51380224);   // 102760448 B (conv1 out, NHWC bf16)

  pack_w_kernel<64> <<<288, 256, 0, stream>>>(w1, W1p);
  pack_w_kernel<128><<<576, 256, 0, stream>>>(w2, W2p);
  to_nhwc_kernel    <<<32 * 112, 256, 0, stream>>>(x, Xh);

  dim3 grid(7, 28, 32);  // (112/16) x (112/4) x B
  conv3x3_wmma_kernel<64,  false><<<grid, 256, 0, stream>>>(Xh, W1p, b1, Y1, nullptr);
  conv3x3_wmma_kernel<128, true ><<<grid, 256, 0, stream>>>(Y1, W2p, b2, nullptr, (float*)d_out);
}
